// MLP_76605036691741
// MI455X (gfx1250) — matrix operand order repaired and checked by reading, not compiled, not run on hardware
//
#include <hip/hip_runtime.h>
#include <hip/hip_bf16.h>

// ---------- types ----------
typedef __attribute__((ext_vector_type(16))) __bf16    v16bf;
typedef __attribute__((ext_vector_type(8)))  float     v8f;
typedef __attribute__((ext_vector_type(4)))  unsigned  vu4;

union Frag16 {           // 32 bytes: one WMMA 16-bit A or B fragment (8 VGPRs)
    vu4   q[2];
    v16bf v;
};

__device__ __forceinline__ unsigned short f2bf(float f) {
    union { float f; unsigned u; } x; x.f = f;
    unsigned r = x.u + 0x7FFFu + ((x.u >> 16) & 1u);   // round-to-nearest-even
    return (unsigned short)(r >> 16);
}
__device__ __forceinline__ float bf2f(unsigned short b) {
    union { unsigned u; float f; } x; x.u = ((unsigned)b) << 16;
    return x.f;
}

// ---------- prep: convert W1 (128x256) and W2 (128x128) to bf16 in workspace ----------
__global__ void convert_weights_kernel(const float* __restrict__ W1,
                                       const float* __restrict__ W2,
                                       unsigned short* __restrict__ w1b,
                                       unsigned short* __restrict__ w2b) {
    int i = blockIdx.x * 256 + threadIdx.x;          // 128 blocks * 256 = 32768
    if (i < 128 * 256) w1b[i] = f2bf(W1[i]);
    if (i < 128 * 128) w2b[i] = f2bf(W2[i]);
}

// ---------- main fused edge-MLP kernel ----------
// block = 256 threads (8 wave32), processes 64 edges.
__global__ __launch_bounds__(256)
void gnn_edge_mlp_kernel(const float* __restrict__ h,
                         const int*   __restrict__ src,
                         const int*   __restrict__ dst,
                         const unsigned short* __restrict__ w1b,   // [128][256] bf16
                         const float* __restrict__ b1,
                         const unsigned short* __restrict__ w2b,   // [128][128] bf16
                         const float* __restrict__ b2,
                         const float* __restrict__ W3,             // [128] f32
                         const float* __restrict__ b3,
                         float* __restrict__ out,
                         int nEdges) {
    // LDS: BUF doubles as (a) gathered [h_u|h_v] bf16 tile 64x256 for layer 1,
    //      (b) layer-2 output X2 64x128 (reused after layer-1 reads complete).
    __shared__ __align__(16) unsigned short BUF[64 * 256];   // 32 KB
    __shared__ __align__(16) unsigned short X1 [64 * 128];   // 16 KB
    __shared__ float red[256];                               // 1 KB, layer-3 partials
    __shared__ int sidx[64];
    __shared__ int didx[64];

    const int t  = threadIdx.x;
    const int e0 = blockIdx.x * 64;

    // ---- load (clamped) edge indices ----
    if (t < 64) {
        int e = e0 + t;
        sidx[t] = src[e < nEdges ? e : 0];
    } else if (t < 128) {
        int e = e0 + (t - 64);
        didx[t - 64] = dst[e < nEdges ? e : 0];
    }
    __syncthreads();

    // ---- gather 64 rows of [h_u | h_v], f32 -> bf16, into BUF (stride 256) ----
    // 4096 float4-units total; 16 per thread.
    #pragma unroll
    for (int i = 0; i < 16; ++i) {
        int idx = t + i * 256;
        int m   = idx >> 6;          // edge row 0..63
        int cu  = idx & 63;          // float4 unit within row (0..31 = u, 32..63 = v)
        int node = (cu < 32) ? sidx[m] : didx[m];
        const float4 f = ((const float4*)(h + (size_t)node * 128))[cu & 31];
        uint2 packed;
        packed.x = (unsigned)f2bf(f.x) | ((unsigned)f2bf(f.y) << 16);
        packed.y = (unsigned)f2bf(f.z) | ((unsigned)f2bf(f.w) << 16);
        *(uint2*)&BUF[m * 256 + cu * 4] = packed;
    }
    __syncthreads();

    const int wave = t >> 5;
    const int lane = t & 31;
    const int nl   = lane & 15;      // A: row M index / B: column N index / D: column N
    const int hh   = lane >> 4;      // lane half selector
    const int n0   = wave * 16;      // this wave's output-feature tile

    // ================= layer 1: X1 = relu([hu|hv] @ W1^T + b1), K = 256 =================
    {
        v8f acc[4] = {};             // 4 M-tiles of 16 edges each
        #pragma unroll
        for (int kc = 0; kc < 8; ++kc) {
            // Batch-load B fragment + all 4 A fragments first -> one clause, one wait,
            // then 4 back-to-back WMMAs (shared B, independent accumulators).
            Frag16 bf, af[4];
            const vu4* wrow = (const vu4*)(w1b + (n0 + nl) * 256 + kc * 32 + hh * 8);
            bf.q[0] = wrow[0];
            bf.q[1] = wrow[2];
            #pragma unroll
            for (int mt = 0; mt < 4; ++mt) {
                const unsigned short* arow = &BUF[(mt * 16 + nl) * 256 + kc * 32 + hh * 8];
                af[mt].q[0] = *(const vu4*)(arow);
                af[mt].q[1] = *(const vu4*)(arow + 16);
            }
            #pragma unroll
            for (int mt = 0; mt < 4; ++mt)
                acc[mt] = __builtin_amdgcn_wmma_f32_16x16x32_bf16(
                              false, af[mt].v, false, bf.v, (short)0, acc[mt], false, false);
        }
        const float bias = b1[n0 + nl];
        #pragma unroll
        for (int mt = 0; mt < 4; ++mt) {
            #pragma unroll
            for (int r = 0; r < 8; ++r) {
                float v = acc[mt][r] + bias;            // D elem r: row = mt*16 + hh*8 + r
                v = v > 0.0f ? v : 0.0f;
                X1[(mt * 16 + hh * 8 + r) * 128 + (n0 + nl)] = f2bf(v);
            }
        }
    }
    __syncthreads();   // X1 ready; BUF reads done -> reuse BUF as X2

    // ================= layer 2: X2 = relu(X1 @ W2^T + b2), K = 128 =================
    {
        v8f acc[4] = {};
        #pragma unroll
        for (int kc = 0; kc < 4; ++kc) {
            Frag16 bf, af[4];
            const vu4* wrow = (const vu4*)(w2b + (n0 + nl) * 128 + kc * 32 + hh * 8);
            bf.q[0] = wrow[0];
            bf.q[1] = wrow[2];
            #pragma unroll
            for (int mt = 0; mt < 4; ++mt) {
                const unsigned short* arow = &X1[(mt * 16 + nl) * 128 + kc * 32 + hh * 8];
                af[mt].q[0] = *(const vu4*)(arow);
                af[mt].q[1] = *(const vu4*)(arow + 16);
            }
            #pragma unroll
            for (int mt = 0; mt < 4; ++mt)
                acc[mt] = __builtin_amdgcn_wmma_f32_16x16x32_bf16(
                              false, af[mt].v, false, bf.v, (short)0, acc[mt], false, false);
        }
        const float bias = b2[n0 + nl];
        #pragma unroll
        for (int mt = 0; mt < 4; ++mt) {
            #pragma unroll
            for (int r = 0; r < 8; ++r) {
                float v = acc[mt][r] + bias;
                v = v > 0.0f ? v : 0.0f;
                BUF[(mt * 16 + hh * 8 + r) * 128 + (n0 + nl)] = f2bf(v);   // X2
            }
        }
    }
    __syncthreads();

    // ================= layer 3: score = X2 @ W3^T + b3 =================
    // 4 threads per edge, 32 features each, LDS tree reduction.
    {
        const int m = t >> 2;        // edge row 0..63
        const int p = t & 3;         // feature quarter
        const unsigned short* xr = &BUF[m * 128 + p * 32];
        const float* wr = W3 + p * 32;
        float s = 0.0f;
        #pragma unroll
        for (int k = 0; k < 32; ++k)
            s += bf2f(xr[k]) * wr[k];
        red[t] = s;
    }
    __syncthreads();
    if (t < 64) {
        float s = b3[0] + red[4 * t] + red[4 * t + 1] + red[4 * t + 2] + red[4 * t + 3];
        int e = e0 + t;
        if (e < nEdges) out[e] = s;
    }
}

// ---------- harness entry ----------
extern "C" void kernel_launch(void* const* d_in, const int* in_sizes, int n_in,
                              void* d_out, int out_size, void* d_ws, size_t ws_size,
                              hipStream_t stream) {
    const float* h   = (const float*)d_in[0];
    const int*   src = (const int*)  d_in[1];
    const int*   dst = (const int*)  d_in[2];
    const float* W1  = (const float*)d_in[3];
    const float* b1  = (const float*)d_in[4];
    const float* W2  = (const float*)d_in[5];
    const float* b2  = (const float*)d_in[6];
    const float* W3  = (const float*)d_in[7];
    const float* b3  = (const float*)d_in[8];
    float* out = (float*)d_out;

    const int nEdges = in_sizes[1];

    unsigned short* w1b = (unsigned short*)d_ws;              // 128*256 bf16 = 64 KB
    unsigned short* w2b = w1b + 128 * 256;                    // 128*128 bf16 = 32 KB

    convert_weights_kernel<<<128, 256, 0, stream>>>(W1, W2, w1b, w2b);

    const int blocks = (nEdges + 63) / 64;
    gnn_edge_mlp_kernel<<<blocks, 256, 0, stream>>>(
        h, src, dst, w1b, b1, w2b, b2, W3, b3, out, nEdges);
}
